// TransformerModel_S_15762529976466
// MI455X (gfx1250) — matrix operand order repaired and checked by reading, not compiled, not run on hardware
//
#include <hip/hip_runtime.h>

// ---------------------------------------------------------------------------
// Transformer encoder-decoder forward on gfx1250 using v_wmma_f32_16x16x32_f16.
// D=128, H=4 (head dim 32), L=3, B=64, S=696, T=512 (decoder length 511).
// GEMM: one wave computes a 16x64 strip (A-fragment reused over 4 WMMAs/k-step).
// Attention: one wave handles 32 query rows (K/V fragments reused over 2 Q tiles).
// ---------------------------------------------------------------------------

#define S_LEN   696
#define T_FULL  512
#define T_DEC   511
#define BATCH   64
#define DMODEL  128
#define NHEAD   4
#define HDIM    32
#define NLAYER  3

#define NE (S_LEN * BATCH)   // 44544 encoder tokens
#define ND (T_DEC * BATCH)   // 32704 decoder tokens

typedef __attribute__((ext_vector_type(16))) _Float16 v16h;
typedef __attribute__((ext_vector_type(8)))  float    v8f;

__device__ __forceinline__ v8f wmma16(v16h a, v16h b, v8f c) {
  // D = A(16x32 f16) * B(32x16 f16) + C(16x16 f32)
  return __builtin_amdgcn_wmma_f32_16x16x32_f16(false, a, false, b, (short)0, c,
                                                false, false);
}
__device__ __forceinline__ int imin(int a, int b) { return a < b ? a : b; }

// A-fragment K index (f16 16x32): VGPR j holds K = ka(j), ka(j)+1
__device__ __forceinline__ int ka_idx(int j, int lane) {
  return 2 * j + (lane & 16);
}

// ---------------------------------------------------------------------------
// GEMM: C[M,N] = A[M,128] @ W[N,128]^T + bias[N]   (optional ReLU)
// One wave computes a 16x64 strip (4 WMMA column tiles); 8 waves per block.
// A fragment loaded once per k-step and reused for all 4 column tiles.
// ---------------------------------------------------------------------------
__global__ __launch_bounds__(256) void gemm128_kernel(
    const float* __restrict__ A, const float* __restrict__ W,
    const float* __restrict__ bias, float* __restrict__ C,
    int M, int N, int relu) {
  int lane = threadIdx.x & 31;
  int wave = threadIdx.x >> 5;
  int ng = N >> 6;                       // 64-wide column groups
  int tiles = ((M + 15) >> 4) * ng;
  int tile = blockIdx.x * 8 + wave;
  if (tile >= tiles) return;             // wave-uniform exit
  int tm = tile / ng, tg = tile % ng;
  int cl = lane & 15;
  int hi = (lane & 16) >> 1;             // 0 or 8

  int arow = imin(tm * 16 + cl, M - 1);
  const float* Ap = A + (size_t)arow * DMODEL;
  const float* Wp = W + (size_t)(tg * 64 + cl) * DMODEL + (lane & 16);

  v8f acc[4];
#pragma unroll
  for (int n = 0; n < 4; ++n) {
    float bv = bias[tg * 64 + n * 16 + cl];
#pragma unroll
    for (int r = 0; r < 8; ++r) acc[n][r] = bv;
  }

#pragma unroll
  for (int k0 = 0; k0 < DMODEL; k0 += 32) {
    v16h af;
#pragma unroll
    for (int j = 0; j < 8; ++j) {
      int k = k0 + 2 * j + 2 * hi;
      af[2 * j]     = (_Float16)Ap[k];
      af[2 * j + 1] = (_Float16)Ap[k + 1];
    }
#pragma unroll
    for (int n = 0; n < 4; ++n) {
      v16h bf;
#pragma unroll
      for (int e = 0; e < 16; ++e)
        bf[e] = (_Float16)Wp[(size_t)n * 16 * DMODEL + k0 + e];
      acc[n] = wmma16(af, bf, acc[n]);
    }
  }

#pragma unroll
  for (int n = 0; n < 4; ++n) {
#pragma unroll
    for (int r = 0; r < 8; ++r) {
      int row = tm * 16 + r + hi;
      if (row < M) {
        float v = acc[n][r];
        if (relu) v = fmaxf(v, 0.f);
        C[(size_t)row * N + tg * 64 + n * 16 + cl] = v;
      }
    }
  }
}

// ---------------------------------------------------------------------------
// Flash attention: one wave per (batch, head, 32-row query block = 2 tiles).
// Token t, batch b lives at row (t*BATCH + b).  Online softmax over 32-wide
// key chunks; K/V fragments shared by both query tiles; P transposed via LDS.
// ---------------------------------------------------------------------------
__global__ __launch_bounds__(32) void attn_kernel(
    const float* __restrict__ Qp, int ldq,
    const float* __restrict__ Kp, int ldk,
    const float* __restrict__ Vp, int ldv,
    float* __restrict__ Op, int ldo,
    int Tq, int Tk, int causal) {
  __shared__ _Float16 sP[32][34];

  int qblocks = (Tq + 31) >> 5;
  int qb = blockIdx.x % qblocks;
  int bh = blockIdx.x / qblocks;
  int h = bh % NHEAD;
  int b = bh / NHEAD;
  int q0 = qb * 32;

  int lane = threadIdx.x;
  int cl = lane & 15;
  int hi = (lane & 16) >> 1;             // row offset in C layout
  const float scale = 0.17677669529663687f;  // 1/sqrt(32)

  // ---- two Q fragments (A layout, K = head dim 32), prescaled ----
  v16h qf0, qf1;
  {
    int r0 = imin(q0 + cl, Tq - 1);
    int r1 = imin(q0 + 16 + cl, Tq - 1);
    const float* qb0 = Qp + ((size_t)r0 * BATCH + b) * ldq + h * HDIM;
    const float* qb1 = Qp + ((size_t)r1 * BATCH + b) * ldq + h * HDIM;
#pragma unroll
    for (int j = 0; j < 8; ++j) {
      int k = ka_idx(j, lane);
      qf0[2 * j]     = (_Float16)(qb0[k] * scale);
      qf0[2 * j + 1] = (_Float16)(qb0[k + 1] * scale);
      qf1[2 * j]     = (_Float16)(qb1[k] * scale);
      qf1[2 * j + 1] = (_Float16)(qb1[k + 1] * scale);
    }
  }

  v8f o00, o01, o10, o11;
  float mi0[8], li0[8], mi1[8], li1[8];
#pragma unroll
  for (int r = 0; r < 8; ++r) {
    o00[r] = 0.f; o01[r] = 0.f; o10[r] = 0.f; o11[r] = 0.f;
    mi0[r] = -1e30f; li0[r] = 0.f; mi1[r] = -1e30f; li1[r] = 0.f;
  }

  int kend = causal ? imin(Tk, q0 + 32) : Tk;
  for (int kc = 0; kc < kend; kc += 32) {
    // ---- K fragments (B layout: lane = key token, elements = head dim) ----
    int t0 = imin(kc + cl, Tk - 1);
    int t1 = imin(kc + 16 + cl, Tk - 1);
    const float* kb0 = Kp + ((size_t)t0 * BATCH + b) * ldk + h * HDIM + (lane & 16);
    const float* kb1 = Kp + ((size_t)t1 * BATCH + b) * ldk + h * HDIM + (lane & 16);
    v16h kf0, kf1;
#pragma unroll
    for (int e = 0; e < 16; ++e) {
      kf0[e] = (_Float16)kb0[e];
      kf1[e] = (_Float16)kb1[e];
    }
    v8f zro;
#pragma unroll
    for (int r = 0; r < 8; ++r) zro[r] = 0.f;
    v8f s00 = wmma16(qf0, kf0, zro);
    v8f s01 = wmma16(qf0, kf1, zro);
    v8f s10 = wmma16(qf1, kf0, zro);
    v8f s11 = wmma16(qf1, kf1, zro);

    // ---- online softmax update for both query tiles ----
    float p00[8], p01[8], p10[8], p11[8];
#pragma unroll
    for (int r = 0; r < 8; ++r) {
      int c0 = kc + cl;
      int c1 = kc + 16 + cl;
      // query tile 0 (rows q0 + r + hi)
      {
        int rowtok = q0 + r + hi;
        bool v0 = (c0 < Tk) && (!causal || c0 <= rowtok);
        bool v1 = (c1 < Tk) && (!causal || c1 <= rowtok);
        float x0 = v0 ? s00[r] : -1e30f;
        float x1 = v1 ? s01[r] : -1e30f;
        float mx = fmaxf(x0, x1);
#pragma unroll
        for (int off = 1; off < 16; off <<= 1)
          mx = fmaxf(mx, __shfl_xor(mx, off, 16));
        float mnew = fmaxf(mi0[r], mx);
        float al = __expf(mi0[r] - mnew);
        float e0 = v0 ? __expf(x0 - mnew) : 0.f;
        float e1 = v1 ? __expf(x1 - mnew) : 0.f;
        float rs = e0 + e1;
#pragma unroll
        for (int off = 1; off < 16; off <<= 1)
          rs += __shfl_xor(rs, off, 16);
        li0[r] = li0[r] * al + rs;
        mi0[r] = mnew;
        p00[r] = e0; p01[r] = e1;
        o00[r] *= al; o01[r] *= al;
      }
      // query tile 1 (rows q0 + 16 + r + hi)
      {
        int rowtok = q0 + 16 + r + hi;
        bool v0 = (c0 < Tk) && (!causal || c0 <= rowtok);
        bool v1 = (c1 < Tk) && (!causal || c1 <= rowtok);
        float x0 = v0 ? s10[r] : -1e30f;
        float x1 = v1 ? s11[r] : -1e30f;
        float mx = fmaxf(x0, x1);
#pragma unroll
        for (int off = 1; off < 16; off <<= 1)
          mx = fmaxf(mx, __shfl_xor(mx, off, 16));
        float mnew = fmaxf(mi1[r], mx);
        float al = __expf(mi1[r] - mnew);
        float e0 = v0 ? __expf(x0 - mnew) : 0.f;
        float e1 = v1 ? __expf(x1 - mnew) : 0.f;
        float rs = e0 + e1;
#pragma unroll
        for (int off = 1; off < 16; off <<= 1)
          rs += __shfl_xor(rs, off, 16);
        li1[r] = li1[r] * al + rs;
        mi1[r] = mnew;
        p10[r] = e0; p11[r] = e1;
        o10[r] *= al; o11[r] *= al;
      }
    }

    // ---- transpose both P tiles from C layout to A layout via LDS ----
    __syncthreads();
#pragma unroll
    for (int r = 0; r < 8; ++r) {
      sP[r + hi][cl]           = (_Float16)p00[r];
      sP[r + hi][16 + cl]      = (_Float16)p01[r];
      sP[16 + r + hi][cl]      = (_Float16)p10[r];
      sP[16 + r + hi][16 + cl] = (_Float16)p11[r];
    }
    __syncthreads();
    v16h pf0, pf1;
#pragma unroll
    for (int j = 0; j < 8; ++j) {
      int k = ka_idx(j, lane);
      pf0[2 * j]     = sP[cl][k];
      pf0[2 * j + 1] = sP[cl][k + 1];
      pf1[2 * j]     = sP[16 + cl][k];
      pf1[2 * j + 1] = sP[16 + cl][k + 1];
    }

    // ---- V fragments (B layout: lane = head col, elements = key token) ----
    v16h vf0, vf1;
#pragma unroll
    for (int e = 0; e < 16; ++e) {
      int tok = imin(kc + e + (lane & 16), Tk - 1);
      const float* vb = Vp + ((size_t)tok * BATCH + b) * ldv + h * HDIM;
      vf0[e] = (_Float16)vb[cl];
      vf1[e] = (_Float16)vb[16 + cl];
    }
    o00 = wmma16(pf0, vf0, o00);
    o01 = wmma16(pf0, vf1, o01);
    o10 = wmma16(pf1, vf0, o10);
    o11 = wmma16(pf1, vf1, o11);
  }

  // ---- epilogue: normalize and store both tiles ----
#pragma unroll
  for (int r = 0; r < 8; ++r) {
    int rt0 = q0 + r + hi;
    if (rt0 < Tq) {
      float inv = 1.f / fmaxf(li0[r], 1e-20f);
      float* ob = Op + ((size_t)rt0 * BATCH + b) * ldo + h * HDIM;
      ob[cl]      = o00[r] * inv;
      ob[16 + cl] = o01[r] * inv;
    }
    int rt1 = q0 + 16 + r + hi;
    if (rt1 < Tq) {
      float inv = 1.f / fmaxf(li1[r], 1e-20f);
      float* ob = Op + ((size_t)rt1 * BATCH + b) * ldo + h * HDIM;
      ob[cl]      = o10[r] * inv;
      ob[16 + cl] = o11[r] * inv;
    }
  }
}

// ---------------------------------------------------------------------------
// x = LayerNorm(x + a) * w + b   (one wave per token, 128 elems)
// ---------------------------------------------------------------------------
__global__ __launch_bounds__(256) void add_ln_kernel(
    float* __restrict__ x, const float* __restrict__ a,
    const float* __restrict__ w, const float* __restrict__ b, int Ntok) {
  int lane = threadIdx.x & 31;
  int wave = threadIdx.x >> 5;
  int tok = blockIdx.x * 8 + wave;
  if (tok >= Ntok) return;
  size_t base = (size_t)tok * DMODEL + lane * 4;
  float v[4];
  float s = 0.f;
#pragma unroll
  for (int i = 0; i < 4; ++i) { v[i] = x[base + i] + a[base + i]; s += v[i]; }
#pragma unroll
  for (int off = 1; off < 32; off <<= 1) s += __shfl_xor(s, off, 32);
  float mean = s * (1.f / 128.f);
  float vs = 0.f;
#pragma unroll
  for (int i = 0; i < 4; ++i) { float d = v[i] - mean; vs += d * d; }
#pragma unroll
  for (int off = 1; off < 32; off <<= 1) vs += __shfl_xor(vs, off, 32);
  float rstd = rsqrtf(vs * (1.f / 128.f) + 1e-5f);
#pragma unroll
  for (int i = 0; i < 4; ++i)
    x[base + i] = (v[i] - mean) * rstd * w[lane * 4 + i] + b[lane * 4 + i];
}

// ---------------------------------------------------------------------------
// Encoder embedding: x[t*B+b][d] = src[b][t]*w[d] + bias[d] + pe(t,d)
// ---------------------------------------------------------------------------
__global__ __launch_bounds__(256) void enc_embed_kernel(
    const float* __restrict__ src, const float* __restrict__ w,
    const float* __restrict__ bias, float* __restrict__ x) {
  int idx = blockIdx.x * 256 + threadIdx.x;
  if (idx >= NE * DMODEL) return;
  int d = idx & (DMODEL - 1);
  int tb = idx >> 7;
  int b = tb % BATCH;
  int t = tb / BATCH;
  float freq = __expf(-(float)(d & ~1) * (9.210340371976184f / (float)DMODEL));
  float ang = (float)t * freq;
  float pe = (d & 1) ? __cosf(ang) : __sinf(ang);
  x[idx] = src[(size_t)b * S_LEN + t] * w[d] + bias[d] + pe;
}

// Decoder embedding: y[t*B+b][d] = tgt[b][t]*w[d] + bias[d] + (t==0 ? pe(0,d) : 0)
__global__ __launch_bounds__(256) void dec_embed_kernel(
    const float* __restrict__ tgt, const float* __restrict__ w,
    const float* __restrict__ bias, float* __restrict__ y) {
  int idx = blockIdx.x * 256 + threadIdx.x;
  if (idx >= ND * DMODEL) return;
  int d = idx & (DMODEL - 1);
  int tb = idx >> 7;
  int b = tb % BATCH;
  int t = tb / BATCH;
  float pe0 = (t == 0) ? ((d & 1) ? 1.f : 0.f) : 0.f;  // pe[0]: sin(0)=0, cos(0)=1
  y[idx] = tgt[(size_t)b * T_FULL + t] * w[d] + bias[d] + pe0;
}

// Head: out[b*T_DEC + t] = dot(y[t*B+b], hw) + hb   (one wave per token)
__global__ __launch_bounds__(256) void head_kernel(
    const float* __restrict__ y, const float* __restrict__ hw,
    const float* __restrict__ hb, float* __restrict__ out) {
  int lane = threadIdx.x & 31;
  int wave = threadIdx.x >> 5;
  int tok = blockIdx.x * 8 + wave;
  if (tok >= ND) return;
  int t = tok / BATCH;
  int b = tok % BATCH;
  size_t base = (size_t)tok * DMODEL + lane * 4;
  float s = 0.f;
#pragma unroll
  for (int i = 0; i < 4; ++i) s += y[base + i] * hw[lane * 4 + i];
#pragma unroll
  for (int off = 1; off < 32; off <<= 1) s += __shfl_xor(s, off, 32);
  if (lane == 0) out[(size_t)b * T_DEC + t] = s + hb[0];
}

// ---------------------------------------------------------------------------
// Host-side sequencing
// ---------------------------------------------------------------------------
static inline void run_gemm(const float* A, const float* W, const float* bias,
                            float* C, int M, int N, int relu, hipStream_t s) {
  int tiles = ((M + 15) / 16) * (N / 64);
  int blocks = (tiles + 7) / 8;
  gemm128_kernel<<<blocks, 256, 0, s>>>(A, W, bias, C, M, N, relu);
}

static inline void run_attn(const float* Q, int ldq, const float* K, int ldk,
                            const float* V, int ldv, float* O, int ldo,
                            int Tq, int Tk, int causal, hipStream_t s) {
  int qblocks = (Tq + 31) / 32;
  attn_kernel<<<BATCH * NHEAD * qblocks, 32, 0, s>>>(Q, ldq, K, ldk, V, ldv,
                                                     O, ldo, Tq, Tk, causal);
}

static inline void run_ln(float* x, const float* a, const float* w,
                          const float* b, int Ntok, hipStream_t s) {
  add_ln_kernel<<<(Ntok + 7) / 8, 256, 0, s>>>(x, a, w, b, Ntok);
}

extern "C" void kernel_launch(void* const* d_in, const int* in_sizes, int n_in,
                              void* d_out, int out_size, void* d_ws, size_t ws_size,
                              hipStream_t stream) {
  (void)in_sizes; (void)n_in; (void)out_size; (void)ws_size;
  const float* src        = (const float*)d_in[0];
  const float* tgt        = (const float*)d_in[1];
  const float* emb_in_w   = (const float*)d_in[2];
  const float* emb_in_b   = (const float*)d_in[3];
  const float* emb_out_w  = (const float*)d_in[4];
  const float* emb_out_b  = (const float*)d_in[5];
  const float* enc_qkv_w  = (const float*)d_in[6];
  const float* enc_qkv_b  = (const float*)d_in[7];
  const float* enc_proj_w = (const float*)d_in[8];
  const float* enc_proj_b = (const float*)d_in[9];
  const float* enc_ff1_w  = (const float*)d_in[10];
  const float* enc_ff1_b  = (const float*)d_in[11];
  const float* enc_ff2_w  = (const float*)d_in[12];
  const float* enc_ff2_b  = (const float*)d_in[13];
  const float* enc_ln_w   = (const float*)d_in[14];
  const float* enc_ln_b   = (const float*)d_in[15];
  const float* dsa_qkv_w  = (const float*)d_in[16];
  const float* dsa_qkv_b  = (const float*)d_in[17];
  const float* dsa_proj_w = (const float*)d_in[18];
  const float* dsa_proj_b = (const float*)d_in[19];
  const float* dca_qkv_w  = (const float*)d_in[20];
  const float* dca_qkv_b  = (const float*)d_in[21];
  const float* dca_proj_w = (const float*)d_in[22];
  const float* dca_proj_b = (const float*)d_in[23];
  const float* dec_ff1_w  = (const float*)d_in[24];
  const float* dec_ff1_b  = (const float*)d_in[25];
  const float* dec_ff2_w  = (const float*)d_in[26];
  const float* dec_ff2_b  = (const float*)d_in[27];
  const float* dec_ln_w   = (const float*)d_in[28];
  const float* dec_ln_b   = (const float*)d_in[29];
  const float* head_w     = (const float*)d_in[30];
  const float* head_b     = (const float*)d_in[31];

  float* ws  = (float*)d_ws;
  float* x   = ws;                               // [NE,128]  encoder state / memory
  float* qkv = x + (size_t)NE * DMODEL;          // [NE,384]  qkv / dec q + cross kv
  float* y   = qkv + (size_t)NE * 3 * DMODEL;    // [ND,128]  decoder state
  float* t1  = y + (size_t)ND * DMODEL;          // [NE,128]  attn out / ff hidden
  float* t2  = t1 + (size_t)NE * DMODEL;         // [NE,128]  sublayer output

  // ---------------- encoder ----------------
  enc_embed_kernel<<<(NE * DMODEL + 255) / 256, 256, 0, stream>>>(
      src, emb_in_w, emb_in_b, x);
  for (int i = 0; i < NLAYER; ++i) {
    run_gemm(x, enc_qkv_w + (size_t)i * 384 * 128, enc_qkv_b + i * 384,
             qkv, NE, 384, 0, stream);
    run_attn(qkv, 384, qkv + 128, 384, qkv + 256, 384, t1, 128,
             S_LEN, S_LEN, 0, stream);
    run_gemm(t1, enc_proj_w + (size_t)i * 128 * 128, enc_proj_b + i * 128,
             t2, NE, 128, 0, stream);
    run_ln(x, t2, enc_ln_w + i * 256, enc_ln_b + i * 256, NE, stream);
    run_gemm(x, enc_ff1_w + (size_t)i * 128 * 128, enc_ff1_b + i * 128,
             t1, NE, 128, 1, stream);
    run_gemm(t1, enc_ff2_w + (size_t)i * 128 * 128, enc_ff2_b + i * 128,
             t2, NE, 128, 0, stream);
    run_ln(x, t2, enc_ln_w + i * 256 + 128, enc_ln_b + i * 256 + 128, NE, stream);
  }

  // ---------------- decoder ----------------
  dec_embed_kernel<<<(ND * DMODEL + 255) / 256, 256, 0, stream>>>(
      tgt, emb_out_w, emb_out_b, y);
  for (int i = 0; i < NLAYER; ++i) {
    // self attention (causal)
    run_gemm(y, dsa_qkv_w + (size_t)i * 384 * 128, dsa_qkv_b + i * 384,
             qkv, ND, 384, 0, stream);
    run_attn(qkv, 384, qkv + 128, 384, qkv + 256, 384, t1, 128,
             T_DEC, T_DEC, 1, stream);
    run_gemm(t1, dsa_proj_w + (size_t)i * 128 * 128, dsa_proj_b + i * 128,
             t2, ND, 128, 0, stream);
    run_ln(y, t2, dec_ln_w + i * 384, dec_ln_b + i * 384, ND, stream);
    // cross attention: Q from y, K/V from encoder memory
    float* cq  = qkv;                      // [ND,128]
    float* ckv = qkv + (size_t)ND * 128;   // [NE,256]  (K | V)
    run_gemm(y, dca_qkv_w + (size_t)i * 384 * 128, dca_qkv_b + i * 384,
             cq, ND, 128, 0, stream);
    run_gemm(x, dca_qkv_w + (size_t)i * 384 * 128 + 128 * 128,
             dca_qkv_b + i * 384 + 128, ckv, NE, 256, 0, stream);
    run_attn(cq, 128, ckv, 256, ckv + 128, 256, t1, 128,
             T_DEC, S_LEN, 0, stream);
    run_gemm(t1, dca_proj_w + (size_t)i * 128 * 128, dca_proj_b + i * 128,
             t2, ND, 128, 0, stream);
    run_ln(y, t2, dec_ln_w + i * 384 + 128, dec_ln_b + i * 384 + 128, ND, stream);
    // feed forward
    run_gemm(y, dec_ff1_w + (size_t)i * 128 * 128, dec_ff1_b + i * 128,
             t1, ND, 128, 1, stream);
    run_gemm(t1, dec_ff2_w + (size_t)i * 128 * 128, dec_ff2_b + i * 128,
             t2, ND, 128, 0, stream);
    run_ln(y, t2, dec_ln_w + i * 384 + 256, dec_ln_b + i * 384 + 256, ND, stream);
  }

  head_kernel<<<(ND + 7) / 8, 256, 0, stream>>>(y, head_w, head_b, (float*)d_out);
}
